// PointNetPlusPlus_79654463471955
// MI455X (gfx1250) — matrix operand order repaired and checked by reading, not compiled, not run on hardware
//
#include <hip/hip_runtime.h>
#include <hip/hip_bf16.h>

// ---------------------------------------------------------------------------
// PointNet++ forward for MI455X (gfx1250, wave32).
// GEMM core uses v_wmma_f32_16x16x32_f16 (f16 in, f32 acc).
// ---------------------------------------------------------------------------

typedef _Float16 v16h __attribute__((ext_vector_type(16)));
typedef float    v8f  __attribute__((ext_vector_type(8)));

union F16Frag { v16h h; unsigned u[8]; };

struct LayerP {
  const _Float16* wt;   // [nout][kpad] f16, transposed + zero padded
  const float* b;
  const float* g;       // gamma (if hasBN)
  const float* be;      // beta  (if hasBN)
  int kpad;             // padded K (multiple of 32)
  int nout;             // multiple of 16
  int hasBN;
};

// ---------------------------------------------------------------------------
// Weight prep: W[kin][nout] f32 -> Wt[nout][kpad] f16 (zero padded)
// ---------------------------------------------------------------------------
__global__ void wprep_kernel(const float* __restrict__ W, int kin, int nout,
                             int kpad, _Float16* __restrict__ out) {
  int e = blockIdx.x * 256 + threadIdx.x;
  if (e >= nout * kpad) return;
  int n = e / kpad;
  int k = e - n * kpad;
  float v = (k < kin) ? W[k * nout + n] : 0.f;
  out[e] = (_Float16)v;
}

// ---------------------------------------------------------------------------
// Farthest point sampling: one persistent block, dmin in registers.
// ---------------------------------------------------------------------------
template <int PTS>
__global__ __launch_bounds__(1024)
void fps_kernel(const float* __restrict__ pos, int stride, int M,
                int* __restrict__ outIdx) {
  const int tid = threadIdx.x;
  const int base = tid * PTS;
  float dm[PTS];
#pragma unroll
  for (int j = 0; j < PTS; ++j) dm[j] = 3.4e38f;

  __shared__ float rv[32];
  __shared__ int ri[32];
  __shared__ int curS;

  int cur = 0;
  if (tid == 0) outIdx[0] = 0;

  for (int s = 1; s < M; ++s) {
    float cx = pos[(long)cur * stride + 0];
    float cy = pos[(long)cur * stride + 1];
    float cz = pos[(long)cur * stride + 2];
    float bv = -1.f;
    int bi = 0x7fffffff;
#pragma unroll
    for (int j = 0; j < PTS; ++j) {
      const float* p = pos + (long)(base + j) * stride;
      float dx = p[0] - cx, dy = p[1] - cy, dz = p[2] - cz;
      float d = dx * dx + dy * dy + dz * dz;
      float nd = fminf(dm[j], d);
      dm[j] = nd;
      if (nd > bv) { bv = nd; bi = base + j; }
    }
    // wave32 reduction (argmax, smallest index on tie)
    for (int sh = 16; sh > 0; sh >>= 1) {
      float v2 = __shfl_xor(bv, sh, 32);
      int i2 = __shfl_xor(bi, sh, 32);
      if (v2 > bv || (v2 == bv && i2 < bi)) { bv = v2; bi = i2; }
    }
    int wid = tid >> 5;
    if ((tid & 31) == 0) { rv[wid] = bv; ri[wid] = bi; }
    __syncthreads();
    if (tid < 32) {
      float v = rv[tid];
      int i = ri[tid];
      for (int sh = 16; sh > 0; sh >>= 1) {
        float v2 = __shfl_xor(v, sh, 32);
        int i2 = __shfl_xor(i, sh, 32);
        if (v2 > v || (v2 == v && i2 < i)) { v = v2; i = i2; }
      }
      if (tid == 0) { outIdx[s] = i; curS = i; }
    }
    __syncthreads();
    cur = curS;
  }
}

// ---------------------------------------------------------------------------
// Gather positions: dst[i][c] = src[idx[i]*stride + c]
// ---------------------------------------------------------------------------
__global__ void gatherpos_kernel(const float* __restrict__ src, int stride,
                                 const int* __restrict__ idx, int M,
                                 float* __restrict__ dst) {
  int e = blockIdx.x * 256 + threadIdx.x;
  if (e >= M * 3) return;
  int i = e / 3;
  int c = e - 3 * i;
  dst[e] = src[(long)idx[i] * stride + c];
}

// ---------------------------------------------------------------------------
// Ball query: per center, compact in-radius candidates to LDS, then extract
// the K nearest by iterative argmin.  One block per center.
// ---------------------------------------------------------------------------
#define RCAP 6144
__global__ __launch_bounds__(256)
void radius_kernel(const float* __restrict__ pos, int posStride, int N,
                   const float* __restrict__ cpos, float r2, int K,
                   int* __restrict__ nbr, int* __restrict__ cntOut) {
  __shared__ float cd[RCAP];
  __shared__ int ci[RCAP];
  __shared__ int cnt;
  __shared__ float rvs[256];
  __shared__ int ris[256];

  const int tid = threadIdx.x;
  const int c = blockIdx.x;
  if (tid == 0) cnt = 0;
  float cx = cpos[c * 3 + 0], cy = cpos[c * 3 + 1], cz = cpos[c * 3 + 2];
  __syncthreads();

  for (int i = tid; i < N; i += 256) {
    if (i + 256 < N) __builtin_prefetch(&pos[(long)(i + 256) * posStride], 0, 1);
    const float* p = pos + (long)i * posStride;
    float dx = p[0] - cx, dy = p[1] - cy, dz = p[2] - cz;
    float d2 = dx * dx + dy * dy + dz * dz;
    if (d2 <= r2) {
      int s = atomicAdd(&cnt, 1);
      if (s < RCAP) { cd[s] = d2; ci[s] = i; }
    }
  }
  __syncthreads();

  int C = cnt < RCAP ? cnt : RCAP;
  int ks = K < C ? K : C;
  for (int k = 0; k < ks; ++k) {
    float bv = 3.4e38f;
    int bs = 0;
    for (int j = tid; j < C; j += 256) {
      float v = cd[j];
      if (v < bv) { bv = v; bs = j; }
    }
    rvs[tid] = bv;
    ris[tid] = bs;
    __syncthreads();
    if (tid < 32) {
      float v = rvs[tid];
      int s = ris[tid];
      for (int o = 32; o < 256; o += 32) {
        float v2 = rvs[tid + o];
        int s2 = ris[tid + o];
        if (v2 < v) { v = v2; s = s2; }
      }
      for (int sh = 16; sh > 0; sh >>= 1) {
        float v2 = __shfl_down(v, sh, 32);
        int s2 = __shfl_down(s, sh, 32);
        if (v2 < v) { v = v2; s = s2; }
      }
      if (tid == 0) {
        nbr[(long)c * K + k] = ci[s];
        cd[s] = 3.4e38f;
      }
    }
    __syncthreads();
  }
  if (tid == 0) cntOut[c] = ks;
  __syncthreads();
  int first = nbr[(long)c * K]; // ks >= 1 (center is in its own ball)
  for (int k = ks + tid; k < K; k += 256) nbr[(long)c * K + k] = first;
}

// ---------------------------------------------------------------------------
// One MLP layer on an LDS row tile via WMMA f16->f32.
//   in : [rows][L.kpad] f16 (LDS), out : [rows][out_ld] f16 (LDS)
// Fragment layouts per CDNA5 ISA 7.12.2.
// ---------------------------------------------------------------------------
__device__ __forceinline__ void run_layer(const _Float16* __restrict__ in,
                                          _Float16* __restrict__ out,
                                          int out_ld, int rows,
                                          const LayerP& L) {
  const int tid = threadIdx.x;
  const int wave = tid >> 5;
  const int lane = tid & 31;
  const int hsel = lane >> 4;
  const int l15 = lane & 15;
  const int mtiles = rows >> 4;
  const int ntiles = L.nout >> 4;
  const int ktiles = L.kpad >> 5;
  const int total = mtiles * ntiles;

  for (int t = wave; t < total; t += 8) {
    const int mt = t / ntiles;
    const int nt = t - mt * ntiles;
    v8f acc = {};
    const int rA = mt * 16 + l15;
    const int rB = nt * 16 + l15;
    for (int kt = 0; kt < ktiles; ++kt) {
      F16Frag a, b;
      // A: lane<16 rows 0..15; VGPR 0..3 -> K pairs (kb..kb+7), 4..7 -> (+16)
      const _Float16* pa = in + rA * L.kpad + kt * 32 + hsel * 8;
      const uint4* pa4 = (const uint4*)pa;
      uint4 qa0 = pa4[0];
      uint4 qa1 = pa4[2];
      a.u[0] = qa0.x; a.u[1] = qa0.y; a.u[2] = qa0.z; a.u[3] = qa0.w;
      a.u[4] = qa1.x; a.u[5] = qa1.y; a.u[6] = qa1.z; a.u[7] = qa1.w;
      // B: col = lane&15; VGPR 0..3 -> K (kb..kb+7), 4..7 -> (+16), kb = hsel*8
      const _Float16* pb = L.wt + (long)rB * L.kpad + kt * 32 + hsel * 8;
      const uint4* pb4 = (const uint4*)pb;
      uint4 qb0 = pb4[0];
      uint4 qb1 = pb4[2];
      b.u[0] = qb0.x; b.u[1] = qb0.y; b.u[2] = qb0.z; b.u[3] = qb0.w;
      b.u[4] = qb1.x; b.u[5] = qb1.y; b.u[6] = qb1.z; b.u[7] = qb1.w;
      acc = __builtin_amdgcn_wmma_f32_16x16x32_f16(
          false, a.h, false, b.h, (short)0, acc, false, false);
    }
    // Epilogue: per-channel bias (+BN+ReLU); channel = n = lane&15 based.
    const int n = nt * 16 + l15;
    float bias = L.b[n];
    float ga = 0.f, be = 0.f;
    if (L.hasBN) { ga = L.g[n]; be = L.be[n]; }
    const int mbase = mt * 16 + hsel * 8;
#pragma unroll
    for (int e = 0; e < 8; ++e) {
      float v = acc[e] + bias;
      if (L.hasBN) {
        v = v * ga + be;
        v = v > 0.f ? v : 0.f;
      }
      out[(mbase + e) * out_ld + n] = (_Float16)v;
    }
  }
}

// ---------------------------------------------------------------------------
// Fused gather -> 3-layer MLP -> aggregation.
// mode 0: one block per center; rows = K neighbors; masked max -> xOut (f16)
// mode 1: one block per row tile; max over tile -> encoded atomicMax in gEnc
// ---------------------------------------------------------------------------
__global__ __launch_bounds__(256)
void fused_mlp_kernel(int mode, int rows, int w0, int w1,
                      const int* __restrict__ nbr, const int* __restrict__ cnt,
                      int K, const float* __restrict__ posSrc, int posStride,
                      const float* __restrict__ cpos,
                      const float* __restrict__ featF32,
                      const _Float16* __restrict__ featF16, int featStride,
                      int F, const float* __restrict__ posIn, LayerP L0,
                      LayerP L1, LayerP L2, _Float16* __restrict__ xOut,
                      unsigned* __restrict__ gEnc) {
  extern __shared__ _Float16 smem[];
  _Float16* bufA = smem;                 // [rows][w0]
  _Float16* bufB = smem + rows * w0;     // [rows][w1]
  __shared__ int nbrS[64];
  __shared__ float relS[192];

  const int tid = threadIdx.x;
  const int kin = L0.kpad;

  if (mode == 0) {
    const int center = blockIdx.x;
    float cx = cpos[center * 3 + 0];
    float cy = cpos[center * 3 + 1];
    float cz = cpos[center * 3 + 2];
    if (tid < rows) {
      int j = nbr[(long)center * K + tid];
      nbrS[tid] = j;
      relS[tid * 3 + 0] = posSrc[(long)j * posStride + 0] - cx;
      relS[tid * 3 + 1] = posSrc[(long)j * posStride + 1] - cy;
      relS[tid * 3 + 2] = posSrc[(long)j * posStride + 2] - cz;
    }
    __syncthreads();
    const int total = rows * kin;
    for (int e = tid; e < total; e += 256) {
      int r = e / kin;
      int k = e - r * kin;
      float v = 0.f;
      if (k < F) {
        v = featF32 ? featF32[(long)nbrS[r] * featStride + k]
                    : (float)featF16[(long)nbrS[r] * featStride + k];
      } else if (k < F + 3) {
        v = relS[r * 3 + (k - F)];
      }
      bufA[e] = (_Float16)v;
    }
  } else {
    const long rowBase = (long)blockIdx.x * rows;
    const int total = rows * kin;
    for (int e = tid; e < total; e += 256) {
      int r = e / kin;
      int k = e - r * kin;
      long i = rowBase + r;
      float v = 0.f;
      if (k < F) v = (float)featF16[i * featStride + k];
      else if (k < F + 3) v = posIn[i * 3 + (k - F)];
      bufA[e] = (_Float16)v;
    }
  }
  __syncthreads();

  run_layer(bufA, bufB, L1.kpad, rows, L0);
  __syncthreads();
  run_layer(bufB, bufA, L2.kpad, rows, L1);
  __syncthreads();
  run_layer(bufA, bufB, L2.nout, rows, L2);
  __syncthreads();

  if (mode == 0) {
    const int center = blockIdx.x;
    int c_ = cnt[center];
    c_ = c_ < 1 ? 1 : (c_ > rows ? rows : c_);
    for (int ch = tid; ch < L2.nout; ch += 256) {
      float m = -3.0e38f;
      for (int r = 0; r < c_; ++r)
        m = fmaxf(m, (float)bufB[r * L2.nout + ch]);
      xOut[(long)center * L2.nout + ch] = (_Float16)m;
    }
  } else {
    for (int ch = tid; ch < L2.nout; ch += 256) {
      float m = -3.0e38f;
      for (int r = 0; r < rows; ++r)
        m = fmaxf(m, (float)bufB[r * L2.nout + ch]);
      unsigned bits = __float_as_uint(m);
      unsigned enc = (bits & 0x80000000u) ? ~bits : (bits | 0x80000000u);
      atomicMax(&gEnc[ch], enc);
    }
  }
}

// ---------------------------------------------------------------------------
__global__ void ginit_kernel(unsigned* __restrict__ g) {
  int i = blockIdx.x * 256 + threadIdx.x;
  if (i < 512) g[i] = 0u;
}

// Head MLP on the single global-max vector + log_softmax.
__global__ __launch_bounds__(256)
void head_kernel(const unsigned* __restrict__ gEnc, const float* __restrict__ W0,
                 const float* __restrict__ b0, const float* __restrict__ g0,
                 const float* __restrict__ be0, const float* __restrict__ W1,
                 const float* __restrict__ b1, const float* __restrict__ g1,
                 const float* __restrict__ be1, const float* __restrict__ W2,
                 const float* __restrict__ b2, float* __restrict__ logits) {
  __shared__ float gv[512], h1[256], h2[128], h3[16], red[2];
  const int tid = threadIdx.x;
  for (int c = tid; c < 512; c += 256) {
    unsigned u = gEnc[c];
    unsigned bits = (u & 0x80000000u) ? (u & 0x7fffffffu) : ~u;
    gv[c] = __uint_as_float(bits);
  }
  __syncthreads();
  {
    int c = tid;
    float s = b0[c];
    for (int k = 0; k < 512; ++k) s += gv[k] * W0[k * 256 + c];
    s = s * g0[c] + be0[c];
    h1[c] = fmaxf(s, 0.f);
  }
  __syncthreads();
  if (tid < 128) {
    int c = tid;
    float s = b1[c];
    for (int k = 0; k < 256; ++k) s += h1[k] * W1[k * 128 + c];
    s = s * g1[c] + be1[c];
    h2[c] = fmaxf(s, 0.f);
  }
  __syncthreads();
  if (tid < 16) {
    int c = tid;
    float s = b2[c];
    for (int k = 0; k < 128; ++k) s += h2[k] * W2[k * 16 + c];
    h3[c] = s;
  }
  __syncthreads();
  if (tid == 0) {
    float m = h3[0];
    for (int i = 1; i < 16; ++i) m = fmaxf(m, h3[i]);
    float sum = 0.f;
    for (int i = 0; i < 16; ++i) sum += expf(h3[i] - m);
    red[0] = m;
    red[1] = logf(sum);
  }
  __syncthreads();
  if (tid < 16) logits[tid] = h3[tid] - red[0] - red[1];
}

__global__ void bcast_kernel(const float* __restrict__ logits,
                             float* __restrict__ out, int total) {
  int i = blockIdx.x * 256 + threadIdx.x;
  if (i < total) out[i] = logits[i & 15];
}

// ---------------------------------------------------------------------------
extern "C" void kernel_launch(void* const* d_in, const int* in_sizes, int n_in,
                              void* d_out, int out_size, void* d_ws,
                              size_t ws_size, hipStream_t stream) {
  (void)in_sizes; (void)n_in; (void)ws_size;
  const int N1 = 32768, M1 = 16384, M2 = 8192, K = 64;

  const float* points = (const float*)d_in[0];

  // ---- workspace layout ----
  size_t off = 0;
  auto take = [&](size_t bytes) -> void* {
    off = (off + 255) & ~(size_t)255;
    void* p = (char*)d_ws + off;
    off += bytes;
    return p;
  };
  int* idx1 = (int*)take((size_t)M1 * 4);
  int* idx2 = (int*)take((size_t)M2 * 4);
  float* pos1 = (float*)take((size_t)M1 * 3 * 4);
  float* pos2 = (float*)take((size_t)M2 * 3 * 4);
  int* nbr1 = (int*)take((size_t)M1 * K * 4);
  int* cnt1 = (int*)take((size_t)M1 * 4);
  int* nbr2 = (int*)take((size_t)M2 * K * 4);
  int* cnt2 = (int*)take((size_t)M2 * 4);
  _Float16* x1 = (_Float16*)take((size_t)M1 * 128 * 2);
  _Float16* x2 = (_Float16*)take((size_t)M2 * 256 * 2);
  unsigned* gEnc = (unsigned*)take(512 * 4);
  float* logits = (float*)take(16 * 4);
  _Float16* w10 = (_Float16*)take((size_t)64 * 32 * 2);
  _Float16* w11 = (_Float16*)take((size_t)64 * 64 * 2);
  _Float16* w12 = (_Float16*)take((size_t)128 * 64 * 2);
  _Float16* w20 = (_Float16*)take((size_t)128 * 160 * 2);
  _Float16* w21 = (_Float16*)take((size_t)128 * 128 * 2);
  _Float16* w22 = (_Float16*)take((size_t)256 * 128 * 2);
  _Float16* w30 = (_Float16*)take((size_t)256 * 288 * 2);
  _Float16* w31 = (_Float16*)take((size_t)256 * 256 * 2);
  _Float16* w32 = (_Float16*)take((size_t)512 * 256 * 2);

  auto cdiv = [](int a, int b) { return (a + b - 1) / b; };

  // ---- weight prep (f32 -> transposed, K-padded f16) ----
  struct WP { const float* W; int kin, nout, kpad; _Float16* out; };
  const WP wps[9] = {
      {(const float*)d_in[1], 6, 64, 32, w10},
      {(const float*)d_in[5], 64, 64, 64, w11},
      {(const float*)d_in[9], 64, 128, 64, w12},
      {(const float*)d_in[11], 131, 128, 160, w20},
      {(const float*)d_in[15], 128, 128, 128, w21},
      {(const float*)d_in[19], 128, 256, 128, w22},
      {(const float*)d_in[21], 259, 256, 288, w30},
      {(const float*)d_in[25], 256, 256, 256, w31},
      {(const float*)d_in[29], 256, 512, 256, w32},
  };
  for (int i = 0; i < 9; ++i) {
    int n = wps[i].nout * wps[i].kpad;
    wprep_kernel<<<cdiv(n, 256), 256, 0, stream>>>(wps[i].W, wps[i].kin,
                                                   wps[i].nout, wps[i].kpad,
                                                   wps[i].out);
  }

  // ---- level 1: FPS, ball query, PointNetConv ----
  fps_kernel<32><<<1, 1024, 0, stream>>>(points, 6, M1, idx1);
  gatherpos_kernel<<<cdiv(M1 * 3, 256), 256, 0, stream>>>(points, 6, idx1, M1,
                                                          pos1);
  radius_kernel<<<M1, 256, 0, stream>>>(points, 6, N1, pos1, 4.0f, K, nbr1,
                                        cnt1);
  LayerP s1L0{w10, (const float*)d_in[2], (const float*)d_in[3],
              (const float*)d_in[4], 32, 64, 1};
  LayerP s1L1{w11, (const float*)d_in[6], (const float*)d_in[7],
              (const float*)d_in[8], 64, 64, 1};
  LayerP s1L2{w12, (const float*)d_in[10], nullptr, nullptr, 64, 128, 0};
  fused_mlp_kernel<<<M1, 256, 64 * (64 + 128) * 2, stream>>>(
      0, 64, 64, 128, nbr1, cnt1, K, points, 6, pos1, points + 3, nullptr, 6,
      3, nullptr, s1L0, s1L1, s1L2, x1, nullptr);

  // ---- level 2 ----
  fps_kernel<16><<<1, 1024, 0, stream>>>(pos1, 3, M2, idx2);
  gatherpos_kernel<<<cdiv(M2 * 3, 256), 256, 0, stream>>>(pos1, 3, idx2, M2,
                                                          pos2);
  radius_kernel<<<M2, 256, 0, stream>>>(pos1, 3, M1, pos2, 16.0f, K, nbr2,
                                        cnt2);
  LayerP s2L0{w20, (const float*)d_in[12], (const float*)d_in[13],
              (const float*)d_in[14], 160, 128, 1};
  LayerP s2L1{w21, (const float*)d_in[16], (const float*)d_in[17],
              (const float*)d_in[18], 128, 128, 1};
  LayerP s2L2{w22, (const float*)d_in[20], nullptr, nullptr, 128, 256, 0};
  fused_mlp_kernel<<<M2, 256, 64 * (160 + 256) * 2, stream>>>(
      0, 64, 160, 256, nbr2, cnt2, K, pos1, 3, pos2, nullptr, x1, 128, 128,
      nullptr, s2L0, s2L1, s2L2, x2, nullptr);

  // ---- sa3 + global max ----
  ginit_kernel<<<2, 256, 0, stream>>>(gEnc);
  LayerP s3L0{w30, (const float*)d_in[22], (const float*)d_in[23],
              (const float*)d_in[24], 288, 256, 1};
  LayerP s3L1{w31, (const float*)d_in[26], (const float*)d_in[27],
              (const float*)d_in[28], 256, 256, 1};
  LayerP s3L2{w32, (const float*)d_in[30], nullptr, nullptr, 256, 512, 0};
  fused_mlp_kernel<<<M2 / 32, 256, 32 * (288 + 512) * 2, stream>>>(
      1, 32, 288, 512, nullptr, nullptr, K, nullptr, 0, nullptr, nullptr, x2,
      256, 256, pos2, s3L0, s3L1, s3L2, nullptr, gEnc);

  // ---- head + log_softmax + broadcast ----
  head_kernel<<<1, 256, 0, stream>>>(
      gEnc, (const float*)d_in[31], (const float*)d_in[32],
      (const float*)d_in[33], (const float*)d_in[34], (const float*)d_in[35],
      (const float*)d_in[36], (const float*)d_in[37], (const float*)d_in[38],
      (const float*)d_in[39], (const float*)d_in[40], logits);
  bcast_kernel<<<cdiv(out_size, 256), 256, 0, stream>>>(logits, (float*)d_out,
                                                        out_size);
}
